// VITForClassification_63410897158399
// MI455X (gfx1250) — matrix operand order repaired and checked by reading, not compiled, not run on hardware
//
#include <hip/hip_runtime.h>
#include <hip/hip_bf16.h>
#include <math.h>

// ---------------- problem constants (from reference) ----------------
#define B_   16
#define Cc   3
#define IMG_ 512
#define Pp   16
#define H_   384
#define NH_  6
#define DH_  64
#define L_   12
#define I_   1536
#define NC_  1000
#define S_   1025            // 1024 patches + cls
#define M_   (B_ * S_)       // 16400 rows flattened over batch
#define QT_  65              // ceil(S/16)
#define KT_  33              // ceil(S/32)

typedef __attribute__((ext_vector_type(16))) __bf16 v16bf;
typedef __attribute__((ext_vector_type(8)))  float  v8f;
typedef int v4i_vec __attribute__((vector_size(16)));   // builtin's pointee type

union FragBF { v16bf v; uint4 q[2]; };

__device__ __forceinline__ __bf16 f2bf(float x) { return (__bf16)x; }

// ---- gfx1250 async global->LDS path (ASYNCcnt), with sync fallback ----
#define GLOBAL_AS __attribute__((address_space(1)))
#define LDS_AS    __attribute__((address_space(3)))

#if __has_builtin(__builtin_amdgcn_global_load_async_to_lds_b128)
#define USE_ASYNC_LDS 1
#else
#define USE_ASYNC_LDS 0
#endif

__device__ __forceinline__ void wait_async_lds() {
#if USE_ASYNC_LDS
#if __has_builtin(__builtin_amdgcn_s_wait_asynccnt)
  __builtin_amdgcn_s_wait_asynccnt(0);
#else
  asm volatile("s_wait_asynccnt 0" ::: "memory");
#endif
#endif
}

#if USE_ASYNC_LDS
__device__ __forceinline__ void async_copy16(const __bf16* g, __bf16* l) {
  __builtin_amdgcn_global_load_async_to_lds_b128(
      (GLOBAL_AS v4i_vec*)g, (LDS_AS v4i_vec*)l, 0, 0);
}
#endif

// =====================================================================
// Weight prep: out[n*K + k] = (bf16) W[k*N + n]   (store B transposed,
// so WMMA B-fragments are contiguous 32B per lane)
// =====================================================================
__global__ void transpose_to_bf16(const float* __restrict__ W,
                                  __bf16* __restrict__ out, int K, int N) {
  size_t idx = (size_t)blockIdx.x * blockDim.x + threadIdx.x;
  if (idx >= (size_t)K * N) return;
  int kk = (int)(idx % K);
  int n  = (int)(idx / K);
  out[idx] = f2bf(W[(size_t)kk * N + n]);
}

// =====================================================================
// Patch embedding + cls token + positional embedding -> h (fp32 [B,S,H])
// =====================================================================
__global__ void patch_embed(const float* __restrict__ x,
                            const float* __restrict__ cw,
                            const float* __restrict__ cb,
                            const float* __restrict__ cls,
                            const float* __restrict__ pos,
                            float* __restrict__ h) {
  size_t idx = (size_t)blockIdx.x * blockDim.x + threadIdx.x;
  if (idx >= (size_t)B_ * S_ * H_) return;
  int o = (int)(idx % H_);
  int s = (int)((idx / H_) % S_);
  int b = (int)(idx / ((size_t)H_ * S_));
  float val;
  if (s == 0) {
    val = cls[o];
  } else {
    int p  = s - 1;
    int ph = p >> 5;         // 32 patches per row
    int pw = p & 31;
    const float* wp = cw + (size_t)o * (Cc * Pp * Pp);
    float sum = cb[o];
    for (int c = 0; c < Cc; ++c)
      for (int pi = 0; pi < Pp; ++pi) {
        const float* xr = x + (((size_t)b * Cc + c) * IMG_ + ph * Pp + pi) * IMG_ + pw * Pp;
        const float* wr = wp + (c * Pp + pi) * Pp;
#pragma unroll
        for (int pj = 0; pj < Pp; ++pj) sum += xr[pj] * wr[pj];
      }
    val = sum;
  }
  h[idx] = val + pos[(size_t)s * H_ + o];
}

// =====================================================================
// LayerNorm: fp32 h row -> bf16 hn row. One 128-thread block per row.
// =====================================================================
__global__ __launch_bounds__(128)
void layernorm_bf16(const float* __restrict__ h, const float* __restrict__ w,
                    const float* __restrict__ b, __bf16* __restrict__ out) {
  __shared__ float red[128];
  int row = blockIdx.x;
  const float* hr = h + (size_t)row * H_;
  float s = 0.f;
  for (int i = threadIdx.x; i < H_; i += 128) s += hr[i];
  red[threadIdx.x] = s; __syncthreads();
  for (int off = 64; off > 0; off >>= 1) {
    if (threadIdx.x < off) red[threadIdx.x] += red[threadIdx.x + off];
    __syncthreads();
  }
  float mu = red[0] * (1.0f / H_);
  __syncthreads();
  float vs = 0.f;
  for (int i = threadIdx.x; i < H_; i += 128) { float d = hr[i] - mu; vs += d * d; }
  red[threadIdx.x] = vs; __syncthreads();
  for (int off = 64; off > 0; off >>= 1) {
    if (threadIdx.x < off) red[threadIdx.x] += red[threadIdx.x + off];
    __syncthreads();
  }
  float rstd = rsqrtf(red[0] * (1.0f / H_) + 1e-5f);
  __bf16* o = out + (size_t)row * H_;
  for (int i = threadIdx.x; i < H_; i += 128)
    o[i] = f2bf((hr[i] - mu) * rstd * w[i] + b[i]);
}

// =====================================================================
// bf16 WMMA GEMM: out[M,N] = act(A[M,K] * Bt[N,K]^T + bias) (+resid)
//   block = 128 threads (4 waves), tile 128x64, k-step 32, double-
//   buffered LDS with async global->LDS staging.
//   Wave w owns rows [32w,32w+32): 2 A-frags x 4 B-frags = 8 WMMA/step.
// =====================================================================
__global__ __launch_bounds__(128)
void gemm_bf16_wmma(const __bf16* __restrict__ A, int lda,
                    const __bf16* __restrict__ Bt,     // [N][K] pre-transposed
                    const float* __restrict__ bias,
                    const float* __restrict__ resid,   // nullable, stride N
                    float* __restrict__ outF,          // nullable
                    __bf16* __restrict__ outB,         // nullable
                    int M, int N, int K, int act) {
  __shared__ __align__(16) __bf16 As[2][128][40];  // +8 pad -> conflict-free frags
  __shared__ __align__(16) __bf16 Bs[2][64][40];

  const int tid   = threadIdx.x;
  const int wave  = tid >> 5;
  const int lane  = tid & 31;
  const int half  = lane >> 4;
  const int l16   = lane & 15;
  const int tileN = blockIdx.x * 64;
  const int tileM = blockIdx.y * 128;

  v8f acc[2][4] = {};

  // Stage one 32-wide k-slab of A (128 rows) and Bt (64 rows) into buffer.
  auto stage = [&](int buf, int k0) {
    int gm = tileM + tid;                    // one A row per thread
    const __bf16* ga = A + (size_t)gm * lda + k0;
    __bf16* la = &As[buf][tid][0];
    int rn = tid >> 1, hc = tid & 1;         // B: 2 threads per row
    const __bf16* gb = Bt + (size_t)(tileN + rn) * K + k0 + hc * 16;
    __bf16* lb = &Bs[buf][rn][hc * 16];
#if USE_ASYNC_LDS
    if (gm < M) {
#pragma unroll
      for (int c = 0; c < 4; ++c) async_copy16(ga + c * 8, la + c * 8);
    }
    async_copy16(gb, lb);
    async_copy16(gb + 8, lb + 8);
#else
    uint4 r0 = make_uint4(0, 0, 0, 0), r1 = r0, r2 = r0, r3 = r0;
    if (gm < M) {
      const uint4* src = (const uint4*)ga;
      r0 = src[0]; r1 = src[1]; r2 = src[2]; r3 = src[3];
    }
    ((uint4*)la)[0] = r0; *(uint4*)(la + 8)  = r1;
    *(uint4*)(la + 16) = r2; *(uint4*)(la + 24) = r3;
    const uint4* bsrc = (const uint4*)gb;
    *(uint4*)lb = bsrc[0]; *(uint4*)(lb + 8) = bsrc[1];
#endif
  };

  stage(0, 0);
  wait_async_lds();
  __syncthreads();

  const int nsteps = K >> 5;
  for (int step = 0; step < nsteps; ++step) {
    const int cur = step & 1;
    if (step + 1 < nsteps) stage(cur ^ 1, (step + 1) << 5);

    // A fragments (16x32 bf16, ISA lane layout): row=l16, K={kb..+7, kb+16..+23}
    FragBF a0, a1;
    {
      int kb = half * 8;
      int ar = wave * 32 + l16;
      a0.q[0] = *(const uint4*)&As[cur][ar][kb];
      a0.q[1] = *(const uint4*)&As[cur][ar][kb + 16];
      a1.q[0] = *(const uint4*)&As[cur][ar + 16][kb];
      a1.q[1] = *(const uint4*)&As[cur][ar + 16][kb + 16];
    }
#pragma unroll
    for (int j = 0; j < 4; ++j) {
      // B fragment (32x16): col=l16 (+16j), K={ks..+7, ks+16..+23}, ks = half*8
      FragBF bfr;
      int brow = j * 16 + l16;
      int ks   = half * 8;
      bfr.q[0] = *(const uint4*)&Bs[cur][brow][ks];
      bfr.q[1] = *(const uint4*)&Bs[cur][brow][ks + 16];
      acc[0][j] = __builtin_amdgcn_wmma_f32_16x16x32_bf16(
          false, a0.v, false, bfr.v, (short)0, acc[0][j], false, false);
      acc[1][j] = __builtin_amdgcn_wmma_f32_16x16x32_bf16(
          false, a1.v, false, bfr.v, (short)0, acc[1][j], false, false);
    }
    wait_async_lds();
    __syncthreads();
  }

  // ---- epilogue: C layout m = r + 8*half, n = l16 ----
#pragma unroll
  for (int i = 0; i < 2; ++i) {
#pragma unroll
    for (int j = 0; j < 4; ++j) {
      int gn = tileN + j * 16 + l16;
      float bsv = bias[gn];
#pragma unroll
      for (int r = 0; r < 8; ++r) {
        int gm = tileM + wave * 32 + i * 16 + r + half * 8;
        if (gm < M) {
          float val = acc[i][j][r] + bsv;
          if (act) {
            float u = val;
            val = 0.5f * u * (1.0f + tanhf(0.7978845608f * (u + 0.044715f * u * u * u)));
          }
          size_t oidx = (size_t)gm * N + gn;
          if (resid) val += resid[oidx];
          if (outF) outF[oidx] = val;
          if (outB) outB[oidx] = f2bf(val);
        }
      }
    }
  }
}

// =====================================================================
// Fused flash attention, one wave per (b, head, 16-query tile).
//   QK^T: 4 WMMA per 32-key tile; online softmax via 16-lane shfl_xor;
//   P restaged to LDS in A-frag layout; PV: 4 WMMA with LDS-transposed V.
// =====================================================================
__global__ __launch_bounds__(32)
void attn_kernel(const __bf16* __restrict__ q, const __bf16* __restrict__ k,
                 const __bf16* __restrict__ v, __bf16* __restrict__ out) {
  __shared__ __align__(16) __bf16 vT[64][40];   // [d][key_local]
  __shared__ __align__(16) __bf16 Pl[16][40];   // [m][key_local]

  int bid = blockIdx.x;
  int qt  = bid % QT_;  bid /= QT_;
  int hh  = bid % NH_;
  int bb  = bid / NH_;
  const int lane = threadIdx.x;
  const int half = lane >> 4;
  const int l16  = lane & 15;

  // ---- Q fragments (16x64 = two 16x32 A-frags), rows beyond S zeroed ----
  FragBF aq0, aq1;
  {
    int qrow = qt * 16 + l16;
    int kb   = half * 8;
    if (qrow < S_) {
      const __bf16* qp = q + ((size_t)bb * S_ + qrow) * H_ + hh * DH_;
      aq0.q[0] = *(const uint4*)(qp + kb);
      aq0.q[1] = *(const uint4*)(qp + kb + 16);
      aq1.q[0] = *(const uint4*)(qp + 32 + kb);
      aq1.q[1] = *(const uint4*)(qp + 32 + kb + 16);
    } else {
      aq0.q[0] = aq0.q[1] = aq1.q[0] = aq1.q[1] = make_uint4(0, 0, 0, 0);
    }
  }

  float mrun[8], lrun[8];
  v8f o[4] = {};
#pragma unroll
  for (int r = 0; r < 8; ++r) { mrun[r] = -3.0e38f; lrun[r] = 0.f; }

  for (int kt = 0; kt < KT_; ++kt) {
    const int key0 = kt * 32;

    // ---- stage V^T tile in LDS: lane owns key row key0+lane ----
    {
      int krow = key0 + lane;
      if (krow < S_) {
        const __bf16* vp = v + ((size_t)bb * S_ + krow) * H_ + hh * DH_;
#pragma unroll
        for (int d = 0; d < 64; ++d) vT[d][lane] = vp[d];
      } else {
#pragma unroll
        for (int d = 0; d < 64; ++d) vT[d][lane] = (__bf16)0.f;
      }
    }

    // ---- scores: S0 (keys key0..+15), S1 (keys key0+16..+31) ----
    v8f s0 = {}, s1 = {};
    {
      int ds = half * 8;
      int kc0 = key0 + l16, kc1 = key0 + 16 + l16;
      FragBF b0a, b0b, b1a, b1b;
      if (kc0 < S_) {
        const __bf16* kp = k + ((size_t)bb * S_ + kc0) * H_ + hh * DH_;
        b0a.q[0] = *(const uint4*)(kp + ds);      b0a.q[1] = *(const uint4*)(kp + ds + 16);
        b0b.q[0] = *(const uint4*)(kp + 32 + ds); b0b.q[1] = *(const uint4*)(kp + 32 + ds + 16);
      } else { b0a.q[0]=b0a.q[1]=b0b.q[0]=b0b.q[1]=make_uint4(0,0,0,0); }
      if (kc1 < S_) {
        const __bf16* kp = k + ((size_t)bb * S_ + kc1) * H_ + hh * DH_;
        b1a.q[0] = *(const uint4*)(kp + ds);      b1a.q[1] = *(const uint4*)(kp + ds + 16);
        b1b.q[0] = *(const uint4*)(kp + 32 + ds); b1b.q[1] = *(const uint4*)(kp + 32 + ds + 16);
      } else { b1a.q[0]=b1a.q[1]=b1b.q[0]=b1b.q[1]=make_uint4(0,0,0,0); }
      s0 = __builtin_amdgcn_wmma_f32_16x16x32_bf16(false, aq0.v, false, b0a.v, (short)0, s0, false, false);
      s0 = __builtin_amdgcn_wmma_f32_16x16x32_bf16(false, aq1.v, false, b0b.v, (short)0, s0, false, false);
      s1 = __builtin_amdgcn_wmma_f32_16x16x32_bf16(false, aq0.v, false, b1a.v, (short)0, s1, false, false);
      s1 = __builtin_amdgcn_wmma_f32_16x16x32_bf16(false, aq1.v, false, b1b.v, (short)0, s1, false, false);
    }

    // ---- online softmax (rows live in 16-lane half-wave groups) ----
    {
      const float scale = 0.125f;  // 1/sqrt(64)
      bool kv0 = (key0 + l16) < S_;
      bool kv1 = (key0 + 16 + l16) < S_;
#pragma unroll
      for (int r = 0; r < 8; ++r) {
        float x0 = kv0 ? s0[r] * scale : -3.0e38f;
        float x1 = kv1 ? s1[r] * scale : -3.0e38f;
        float rowmax = fmaxf(x0, x1);
#pragma unroll
        for (int msk = 1; msk < 16; msk <<= 1)
          rowmax = fmaxf(rowmax, __shfl_xor(rowmax, msk, 32));
        float mnew  = fmaxf(mrun[r], rowmax);
        float alpha = __expf(mrun[r] - mnew);
        float p0 = __expf(x0 - mnew);
        float p1 = __expf(x1 - mnew);
        float rs = p0 + p1;
#pragma unroll
        for (int msk = 1; msk < 16; msk <<= 1)
          rs += __shfl_xor(rs, msk, 32);
        lrun[r] = lrun[r] * alpha + rs;
        mrun[r] = mnew;
#pragma unroll
        for (int j = 0; j < 4; ++j) o[j][r] *= alpha;
        s0[r] = p0; s1[r] = p1;
      }
    }

    __syncthreads();  // vT staged; P buffer free to write
    // ---- restage P (C-layout) into LDS in A-frag-compatible row-major ----
#pragma unroll
    for (int r = 0; r < 8; ++r) {
      int m = r + half * 8;
      Pl[m][l16]      = f2bf(s0[r]);
      Pl[m][16 + l16] = f2bf(s1[r]);
    }
    __syncthreads();

    // ---- PV: A = P (16x32), B = V tile via vT (32 keys x 16 d per acc) ----
    FragBF pa;
    {
      int kb = half * 8;
      pa.q[0] = *(const uint4*)&Pl[l16][kb];
      pa.q[1] = *(const uint4*)&Pl[l16][kb + 16];
    }
#pragma unroll
    for (int j = 0; j < 4; ++j) {
      FragBF bv;
      int drow = j * 16 + l16;
      int ks   = half * 8;
      bv.q[0] = *(const uint4*)&vT[drow][ks];
      bv.q[1] = *(const uint4*)&vT[drow][ks + 16];
      o[j] = __builtin_amdgcn_wmma_f32_16x16x32_bf16(
          false, pa.v, false, bv.v, (short)0, o[j], false, false);
    }
    __syncthreads();  // protect vT/Pl before next tile overwrites
  }

  // ---- normalize and write bf16 output [B,S,H] ----
#pragma unroll
  for (int j = 0; j < 4; ++j) {
#pragma unroll
    for (int r = 0; r < 8; ++r) {
      int m    = r + half * 8;
      int qrow = qt * 16 + m;
      if (qrow < S_) {
        float val = o[j][r] / lrun[r];
        out[((size_t)bb * S_ + qrow) * H_ + hh * DH_ + j * 16 + l16] = f2bf(val);
      }
    }
  }
}

// =====================================================================
// Classifier head: logits[b,n] = h[b, cls] . wc[:,n] + bc[n]  (fp32)
// =====================================================================
__global__ void classifier(const float* __restrict__ h, const float* __restrict__ wc,
                           const float* __restrict__ bc, float* __restrict__ outp) {
  int idx = blockIdx.x * blockDim.x + threadIdx.x;
  if (idx >= B_ * NC_) return;
  int n = idx % NC_;
  int b = idx / NC_;
  const float* hr = h + (size_t)b * S_ * H_;
  float sum = bc[n];
  for (int kk = 0; kk < H_; ++kk) sum += hr[kk] * wc[(size_t)kk * NC_ + n];
  outp[idx] = sum;
}

// =====================================================================
// Host orchestration
// =====================================================================
extern "C" void kernel_launch(void* const* d_in, const int* in_sizes, int n_in,
                              void* d_out, int out_size, void* d_ws, size_t ws_size,
                              hipStream_t stream) {
  const float* x      = (const float*)d_in[0];
  const float* conv_w = (const float*)d_in[1];
  const float* conv_b = (const float*)d_in[2];
  const float* cls    = (const float*)d_in[3];
  const float* pos    = (const float*)d_in[4];
  const float* ln1_w  = (const float*)d_in[5];
  const float* ln1_b  = (const float*)d_in[6];
  const float* wq     = (const float*)d_in[7];
  const float* bq     = (const float*)d_in[8];
  const float* wk     = (const float*)d_in[9];
  const float* bk     = (const float*)d_in[10];
  const float* wv     = (const float*)d_in[11];
  const float* bv     = (const float*)d_in[12];
  const float* wo     = (const float*)d_in[13];
  const float* bo     = (const float*)d_in[14];
  const float* ln2_w  = (const float*)d_in[15];
  const float* ln2_b  = (const float*)d_in[16];
  const float* w1     = (const float*)d_in[17];
  const float* b1     = (const float*)d_in[18];
  const float* w2     = (const float*)d_in[19];
  const float* b2     = (const float*)d_in[20];
  const float* wc     = (const float*)d_in[21];
  const float* bc     = (const float*)d_in[22];

  // ---- workspace bump allocator ----
  char* p = (char*)d_ws;
  auto alloc = [&](size_t bytes) -> void* {
    void* r = (void*)p;
    p += (bytes + 255) & ~(size_t)255;
    return r;
  };
  const size_t HH = (size_t)H_ * H_;
  const size_t HI = (size_t)H_ * I_;
  __bf16* wqT  = (__bf16*)alloc(L_ * HH * 2);
  __bf16* wkT  = (__bf16*)alloc(L_ * HH * 2);
  __bf16* wvT  = (__bf16*)alloc(L_ * HH * 2);
  __bf16* woT  = (__bf16*)alloc(L_ * HH * 2);
  __bf16* w1T  = (__bf16*)alloc(L_ * HI * 2);
  __bf16* w2T  = (__bf16*)alloc(L_ * HI * 2);
  float*  h    = (float*) alloc((size_t)M_ * H_ * 4);
  __bf16* hn   = (__bf16*)alloc((size_t)M_ * H_ * 2);
  __bf16* qb   = (__bf16*)alloc((size_t)M_ * H_ * 2);
  __bf16* kb_  = (__bf16*)alloc((size_t)M_ * H_ * 2);
  __bf16* vb   = (__bf16*)alloc((size_t)M_ * H_ * 2);
  __bf16* ab   = (__bf16*)alloc((size_t)M_ * H_ * 2);
  __bf16* actb = (__bf16*)alloc((size_t)M_ * I_ * 2);

  // ---- weight prep: bf16 + transpose to [N][K] ----
  {
    int gHH = (int)((HH + 255) / 256);
    int gHI = (int)((HI + 255) / 256);
    for (int l = 0; l < L_; ++l) {
      transpose_to_bf16<<<gHH, 256, 0, stream>>>(wq + l * HH, wqT + l * HH, H_, H_);
      transpose_to_bf16<<<gHH, 256, 0, stream>>>(wk + l * HH, wkT + l * HH, H_, H_);
      transpose_to_bf16<<<gHH, 256, 0, stream>>>(wv + l * HH, wvT + l * HH, H_, H_);
      transpose_to_bf16<<<gHH, 256, 0, stream>>>(wo + l * HH, woT + l * HH, H_, H_);
      transpose_to_bf16<<<gHI, 256, 0, stream>>>(w1 + l * HI, w1T + l * HI, H_, I_);
      transpose_to_bf16<<<gHI, 256, 0, stream>>>(w2 + l * HI, w2T + l * HI, I_, H_);
    }
  }

  // ---- patch embed ----
  {
    size_t tot = (size_t)B_ * S_ * H_;
    patch_embed<<<(int)((tot + 255) / 256), 256, 0, stream>>>(x, conv_w, conv_b, cls, pos, h);
  }

  // ---- transformer layers ----
  dim3 g384((H_ + 63) / 64, (M_ + 127) / 128);
  dim3 g1536((I_ + 63) / 64, (M_ + 127) / 128);
  for (int l = 0; l < L_; ++l) {
    layernorm_bf16<<<M_, 128, 0, stream>>>(h, ln1_w + l * H_, ln1_b + l * H_, hn);
    gemm_bf16_wmma<<<g384, 128, 0, stream>>>(hn, H_, wqT + l * HH, bq + l * H_,
                                             nullptr, nullptr, qb, M_, H_, H_, 0);
    gemm_bf16_wmma<<<g384, 128, 0, stream>>>(hn, H_, wkT + l * HH, bk + l * H_,
                                             nullptr, nullptr, kb_, M_, H_, H_, 0);
    gemm_bf16_wmma<<<g384, 128, 0, stream>>>(hn, H_, wvT + l * HH, bv + l * H_,
                                             nullptr, nullptr, vb, M_, H_, H_, 0);
    attn_kernel<<<B_ * NH_ * QT_, 32, 0, stream>>>(qb, kb_, vb, ab);
    gemm_bf16_wmma<<<g384, 128, 0, stream>>>(ab, H_, woT + l * HH, bo + l * H_,
                                             h, h, nullptr, M_, H_, H_, 0);
    layernorm_bf16<<<M_, 128, 0, stream>>>(h, ln2_w + l * H_, ln2_b + l * H_, hn);
    gemm_bf16_wmma<<<g1536, 128, 0, stream>>>(hn, H_, w1T + l * HI, b1 + l * I_,
                                              nullptr, nullptr, actb, M_, I_, H_, 1);
    gemm_bf16_wmma<<<g384, 128, 0, stream>>>(actb, I_, w2T + l * HI, b2 + l * H_,
                                             h, h, nullptr, M_, H_, I_, 0);
  }

  // ---- classifier head ----
  classifier<<<(B_ * NC_ + 255) / 256, 256, 0, stream>>>(h, wc, bc, (float*)d_out);
}
